// AtomDecoderLayer_32547262169794
// MI455X (gfx1250) — matrix operand order repaired and checked by reading, not compiled, not run on hardware
//
#include <hip/hip_runtime.h>
#include <hip/hip_bf16.h>
#include <math.h>

// ---------------------------------------------------------------------------
// AtomDecoderLayer for MI455X (gfx1250, wave32, WMMA).
// GEMMs: v_wmma_f32_16x16x32_bf16, 64x128 block tile, 8 waves, each wave
// computes 16x64 (1 A frag x 4 B frags -> 4 accumulators).
// Interior tiles use vectorized b128 staging; only the N=16 GEMM takes the
// guarded path. Async global->LDS copies used in gat_agg (builtin form) and
// pool (inline-asm form).
// ---------------------------------------------------------------------------

typedef __attribute__((ext_vector_type(16))) __bf16 v16bf;
typedef __attribute__((ext_vector_type(8)))  float  v8f;
typedef __attribute__((ext_vector_type(4)))  int    i32x4;

#define LN_EPS    1e-5f
#define NEG_SLOPE 0.01f

enum { EPI_NONE = 0, EPI_BIAS = 1, EPI_BIAS_GELU = 2, EPI_BIAS_SIGMOID = 3, EPI_LEAKY = 4 };

__device__ __forceinline__ unsigned short f2bf(float f) {
    unsigned u = __float_as_uint(f);
    u += 0x7FFFu + ((u >> 16) & 1u);          // round-to-nearest-even
    return (unsigned short)(u >> 16);
}
__device__ __forceinline__ unsigned pack_bf2(float lo, float hi) {
    return (unsigned)f2bf(lo) | ((unsigned)f2bf(hi) << 16);
}

union Frag { v16bf v; unsigned u[8]; };
union Acc  { v8f  v; float    f[8]; };

template <int EPI>
__device__ __forceinline__ float apply_epi(float x, float b) {
    if (EPI == EPI_BIAS)         return x + b;
    if (EPI == EPI_BIAS_GELU)  { float t = x + b; return 0.5f * t * (1.0f + erff(t * 0.70710678118654752f)); }
    if (EPI == EPI_BIAS_SIGMOID){ float t = x + b; return 1.0f / (1.0f + __expf(-t)); }
    if (EPI == EPI_LEAKY)        return x > 0.0f ? x : NEG_SLOPE * x;
    return x;
}

// ---------------------------------------------------------------------------
// Generic GEMM: C[M,N] = epi(A[M,K] @ B[K,N] + bias[N])
// block = 256 threads (8 wave32), block tile 64(m) x 128(n), K-step 32.
// ---------------------------------------------------------------------------
template <int EPI>
__global__ __launch_bounds__(256) void gemm_wmma(
    const float* __restrict__ A, const float* __restrict__ Bm,
    const float* __restrict__ bias, float* __restrict__ C,
    int M, int N, int K)
{
    // stride 36 ushorts = 72B: 8B-aligned rows AND conflict-free fragment
    // gathers (row bank offsets m*18 mod 64 all distinct for m=0..15).
    __shared__ unsigned short Als[64][36];    // [m][k]  bf16
    __shared__ unsigned short Bls[128][36];   // [n][k]  bf16 (B transposed)

    const int tid   = threadIdx.x;
    const int lane  = tid & 31;
    const int wave  = tid >> 5;
    const int m0    = blockIdx.y * 64;
    const int n0    = blockIdx.x * 128;
    const int mi    = (wave & 3) * 16;        // wave m subtile
    const int nbase = (wave >> 2) * 64;       // wave n range: 4 subtiles of 16
    const int half  = lane >> 4;
    const int l16   = lane & 15;

    const bool interior = (m0 + 64 <= M) && (n0 + 128 <= N);

    Acc acc[4];
#pragma unroll
    for (int s = 0; s < 4; s++)
#pragma unroll
        for (int i = 0; i < 8; i++) acc[s].f[i] = 0.f;

    for (int kk = 0; kk < K; kk += 32) {
        if (interior) {
            // ---- A: 64x32 floats = 512 float4, 2 per thread ----
#pragma unroll
            for (int i = 0; i < 2; i++) {
                int e  = tid + i * 256;
                int r  = e >> 3;
                int c4 = (e & 7) << 2;
                float4 va = *(const float4*)&A[(size_t)(m0 + r) * K + kk + c4];
                *(unsigned*)&Als[r][c4]     = pack_bf2(va.x, va.y);
                *(unsigned*)&Als[r][c4 + 2] = pack_bf2(va.z, va.w);
            }
            // ---- B: 32x128 floats = 1024 float4 (coalesced on n), 4/thread ----
#pragma unroll
            for (int i = 0; i < 4; i++) {
                int e  = tid + i * 256;
                int k  = e >> 5;
                int n4 = (e & 31) << 2;
                float4 vb = *(const float4*)&Bm[(size_t)(kk + k) * N + n0 + n4];
                Bls[n4 + 0][k] = f2bf(vb.x);
                Bls[n4 + 1][k] = f2bf(vb.y);
                Bls[n4 + 2][k] = f2bf(vb.z);
                Bls[n4 + 3][k] = f2bf(vb.w);
            }
            // prefetch next K tile (global_prefetch_b8)
            if (kk + 32 < K) {
                __builtin_prefetch(&A[(size_t)(m0 + (tid >> 3)) * K + kk + 32 + ((tid & 7) << 2)], 0, 3);
                __builtin_prefetch(&Bm[(size_t)(kk + 32 + (tid >> 5)) * N + n0 + ((tid & 31) << 2)], 0, 3);
            }
        } else {
            // ---- guarded scalar staging (only the N=16 GEMM lands here) ----
#pragma unroll
            for (int i = 0; i < 8; i++) {
                int e = tid + i * 256;
                int r = e >> 5, c = e & 31;
                float av = 0.f;
                if (m0 + r < M) av = A[(size_t)(m0 + r) * K + kk + c];
                Als[r][c] = f2bf(av);
            }
#pragma unroll
            for (int i = 0; i < 16; i++) {
                int e = tid + i * 256;
                int r = e >> 5, c = e & 31;
                float bv = 0.f;
                if (n0 + r < N) bv = Bm[(size_t)(kk + c) * N + n0 + r];
                Bls[r][c] = f2bf(bv);
            }
        }
        __syncthreads();

        // fragments per ISA 7.12.2 (wave32)
        Frag af, bfr[4];
#pragma unroll
        for (int v = 0; v < 8; v++) {
            int ka = ((v >> 2) << 4) + (half << 3) + ((v & 3) << 1);   // A 16x32
            af.u[v] = *(const unsigned*)&Als[mi + l16][ka];
            int kb = ((v >> 2) << 4) + (half << 3) + ((v & 3) << 1);   // B 32x16
#pragma unroll
            for (int s = 0; s < 4; s++)
                bfr[s].u[v] = *(const unsigned*)&Bls[nbase + s * 16 + l16][kb];
        }
#pragma unroll
        for (int s = 0; s < 4; s++)
            acc[s].v = __builtin_amdgcn_wmma_f32_16x16x32_bf16(false, af.v, false, bfr[s].v,
                                                               (short)0, acc[s].v, false, false);
        __syncthreads();
    }

    // epilogue + store (C/D: vgpr r -> row r + half*8, col = lane&15)
    if (interior) {
#pragma unroll
        for (int s = 0; s < 4; s++) {
            int col = n0 + nbase + s * 16 + l16;
            float b = (EPI == EPI_NONE || EPI == EPI_LEAKY) ? 0.f : bias[col];
#pragma unroll
            for (int r = 0; r < 8; r++) {
                int row = m0 + mi + r + half * 8;
                C[(size_t)row * N + col] = apply_epi<EPI>(acc[s].f[r], b);
            }
        }
    } else {
#pragma unroll
        for (int s = 0; s < 4; s++) {
            int col = n0 + nbase + s * 16 + l16;
            if (col < N) {
                float b = (EPI == EPI_NONE || EPI == EPI_LEAKY) ? 0.f : bias[col];
#pragma unroll
                for (int r = 0; r < 8; r++) {
                    int row = m0 + mi + r + half * 8;
                    if (row < M) C[(size_t)row * N + col] = apply_epi<EPI>(acc[s].f[r], b);
                }
            }
        }
    }
}

// ---------------------------------------------------------------------------
// msg[e,c] += node_msg[gather(e),c] * mask_bw[e] + node_tgt[bn,c]
// ---------------------------------------------------------------------------
__global__ __launch_bounds__(256) void edge_combine_k(
    float* __restrict__ msg, const float* __restrict__ node_msg,
    const float* __restrict__ node_tgt, const int* __restrict__ edge_index,
    const float* __restrict__ mask_bw)
{
    int e  = blockIdx.x;        // bn*32 + ki
    int c  = threadIdx.x;       // 0..255
    int bn = e >> 5;
    int bi = bn >> 10;
    int ei = edge_index[e];
    float mw = mask_bw[e];
    size_t gi = ((size_t)(bi * 1024 + ei)) * 256 + c;
    msg[(size_t)e * 256 + c] += node_msg[gi] * mw + node_tgt[(size_t)bn * 256 + c];
}

// LayerNorm over last dim C, one block per row
__global__ __launch_bounds__(256) void layernorm_k(
    const float* __restrict__ x, const float* __restrict__ g,
    const float* __restrict__ b, float* __restrict__ y, int C)
{
    __shared__ float s1[256], s2[256];
    int row = blockIdx.x;
    const float* xr = x + (size_t)row * C;
    float sum = 0.f, sq = 0.f;
    for (int c = threadIdx.x; c < C; c += 256) { float v = xr[c]; sum += v; sq += v * v; }
    s1[threadIdx.x] = sum; s2[threadIdx.x] = sq;
    __syncthreads();
    for (int s = 128; s > 0; s >>= 1) {
        if (threadIdx.x < s) { s1[threadIdx.x] += s1[threadIdx.x + s]; s2[threadIdx.x] += s2[threadIdx.x + s]; }
        __syncthreads();
    }
    float mean = s1[0] / (float)C;
    float var  = s2[0] / (float)C - mean * mean;
    float inv  = rsqrtf(var + LN_EPS);
    float* yr = y + (size_t)row * C;
    for (int c = threadIdx.x; c < C; c += 256)
        yr[c] = (xr[c] - mean) * inv * g[c] + b[c];
}

__global__ __launch_bounds__(256) void mask_msg_k(float* __restrict__ msg, const int* __restrict__ em)
{
    int e = blockIdx.x, c = threadIdx.x;
    if (em[e] == 0) msg[(size_t)e * 256 + c] = 0.f;
}

// masked mean pool over k, then sigmoid-gate.
// edge_mask row staged into LDS via async copy (inline-asm form).
__global__ __launch_bounds__(256) void pool_k(
    const float* __restrict__ msg, const int* __restrict__ em,
    const float* __restrict__ gate_sig, float* __restrict__ o)
{
    __shared__ int sem[32];
    int bn = blockIdx.x, c = threadIdx.x;
#if defined(__gfx1250__)
    if (threadIdx.x < 8) {
        const int* g = em + bn * 32 + threadIdx.x * 4;
        unsigned lds = (unsigned)(size_t)(__attribute__((address_space(3))) int*)&sem[threadIdx.x * 4];
        asm volatile("global_load_async_to_lds_b128 %0, %1, off"
                     :: "v"(lds), "v"(g) : "memory");
    }
    asm volatile("s_wait_asynccnt 0x0" ::: "memory");
#else
    if (threadIdx.x < 32) sem[threadIdx.x] = em[bn * 32 + threadIdx.x];
#endif
    __syncthreads();
    int scount = 0;
#pragma unroll
    for (int ki = 0; ki < 32; ki++) scount += sem[ki];
    float sden = 1.0f / ((float)scount + 1e-6f);
    float acc = 0.f;
    for (int ki = 0; ki < 32; ki++)
        acc += msg[((size_t)bn * 32 + ki) * 256 + c];   // masked rows already zeroed
    o[(size_t)bn * 256 + c] = acc * sden * gate_sig[(size_t)bn * 256 + c];
}

// per-(node, head) masked softmax over k=32: one wave32 per group
__global__ __launch_bounds__(256) void softmax_k(float* __restrict__ ab, const int* __restrict__ em)
{
    int g    = blockIdx.x * 8 + (threadIdx.x >> 5);   // group = bn*16 + h
    int lane = threadIdx.x & 31;                      // lane = ki
    int bn = g >> 4, h = g & 15;
    size_t idx = ((size_t)bn * 32 + lane) * 16 + h;
    float a = em[bn * 32 + lane] ? ab[idx] : -3.402823466e38f;
    float m = a;
    for (int o = 16; o > 0; o >>= 1) m = fmaxf(m, __shfl_xor(m, o, 32));
    float e = __expf(a - m);
    float s = e;
    for (int o = 16; o > 0; o >>= 1) s += __shfl_xor(s, o, 32);
    ab[idx] = e / s;
}

// og[bn, h*64+d] = sum_k attn[bn,k,h] * v[bn,k,h*64+d]; gated
// attention tile staged into LDS via gfx1250 async global->LDS copy (builtin).
__global__ __launch_bounds__(256) void gat_agg_k(
    const float* __restrict__ attn, const float* __restrict__ v,
    const float* __restrict__ gat_sig, float* __restrict__ og)
{
    __shared__ float sa[32 * 16];
    int bn = blockIdx.x;
#if defined(__gfx1250__) && __has_builtin(__builtin_amdgcn_global_load_async_to_lds_b128)
    if (threadIdx.x < 128) {
        const float* g = attn + (size_t)bn * 512 + threadIdx.x * 4;
        __builtin_amdgcn_global_load_async_to_lds_b128(
            (i32x4*)(void*)g,
            (i32x4*)(void*)&sa[threadIdx.x * 4],
            0, 0);
    }
#if __has_builtin(__builtin_amdgcn_s_wait_asynccnt)
    __builtin_amdgcn_s_wait_asynccnt(0);
#else
    asm volatile("s_wait_asynccnt 0x0" ::: "memory");
#endif
#else
    for (int i = threadIdx.x; i < 512; i += 256)
        sa[i] = attn[(size_t)bn * 512 + i];
#endif
    __syncthreads();
    for (int j = threadIdx.x; j < 1024; j += 256) {
        int h = j >> 6;
        float acc = 0.f;
        for (int ki = 0; ki < 32; ki++)
            acc += sa[ki * 16 + h] * v[((size_t)bn * 32 + ki) * 1024 + j];
        og[(size_t)bn * 1024 + j] = acc * gat_sig[(size_t)bn * 1024 + j];
    }
}

__global__ __launch_bounds__(256) void add3_k(const float* a, const float* b, const float* c,
                                              float* o, int n)
{
    int i = blockIdx.x * 256 + threadIdx.x;
    if (i < n) o[i] = a[i] + b[i] + c[i];
}

__global__ __launch_bounds__(256) void add2_k(const float* a, const float* b, float* o, int n)
{
    int i = blockIdx.x * 256 + threadIdx.x;
    if (i < n) o[i] = a[i] + b[i];
}

// ---------------------------------------------------------------------------
extern "C" void kernel_launch(void* const* d_in, const int* in_sizes, int n_in,
                              void* d_out, int out_size, void* d_ws, size_t ws_size,
                              hipStream_t stream)
{
    (void)in_sizes; (void)n_in; (void)out_size; (void)ws_size;

    const float* node_repr  = (const float*)d_in[0];
    const float* edge_repr  = (const float*)d_in[1];
    const int*   edge_index = (const int*)  d_in[2];
    const int*   edge_mask  = (const int*)  d_in[3];
    const float* mask_bw    = (const float*)d_in[4];
    const float* W_edge_msg = (const float*)d_in[5];
    const float* W_node_src = (const float*)d_in[6];
    const float* W_node_tgt = (const float*)d_in[7];
    const float* msg_ln_g   = (const float*)d_in[8];
    const float* msg_ln_b   = (const float*)d_in[9];
    const float* msg_W1     = (const float*)d_in[10];
    const float* msg_b1     = (const float*)d_in[11];
    const float* msg_W2     = (const float*)d_in[12];
    const float* msg_b2     = (const float*)d_in[13];
    const float* W_gate     = (const float*)d_in[14];
    const float* b_gate     = (const float*)d_in[15];
    const float* W_out      = (const float*)d_in[16];
    const float* W_msg      = (const float*)d_in[17];
    const float* W_attn_bias= (const float*)d_in[18];
    const float* W_gat_value= (const float*)d_in[19];
    const float* W_gat_gate = (const float*)d_in[20];
    const float* b_gat_gate = (const float*)d_in[21];
    const float* W_gat_out  = (const float*)d_in[22];
    const float* node_ln_g  = (const float*)d_in[23];
    const float* node_ln_b  = (const float*)d_in[24];
    const float* node_W1    = (const float*)d_in[25];
    const float* node_b1    = (const float*)d_in[26];
    const float* node_W2    = (const float*)d_in[27];
    const float* node_b2    = (const float*)d_in[28];
    float* out = (float*)d_out;

    const int DIM = 512, MSG = 256, HD = 1024, HID_MSG = 1024, HID_NODE = 2048, Hh = 16;
    const int BN  = 2 * 1024;          // 2048 node rows
    const int E   = BN * 32;           // 65536 edge rows

    size_t off = 0;
    auto alloc = [&](size_t nfloat) -> float* {
        float* p = (float*)((char*)d_ws + off);
        off += ((nfloat * sizeof(float) + 255) & ~(size_t)255);
        return p;
    };

    float* node_msg_b = alloc((size_t)BN * MSG);
    float* node_tgt_b = alloc((size_t)BN * MSG);
    float* gate_sig   = alloc((size_t)BN * MSG);
    float* gat_sig    = alloc((size_t)BN * HD);
    float* o_b        = alloc((size_t)BN * MSG);
    float* dh1        = alloc((size_t)BN * DIM);
    float* og_b       = alloc((size_t)BN * HD);
    float* dh2        = alloc((size_t)BN * DIM);
    float* x_b        = alloc((size_t)BN * DIM);
    float* ln_node    = alloc((size_t)BN * DIM);
    float* node_h     = alloc((size_t)BN * HID_NODE);
    float* mlp_out    = alloc((size_t)BN * DIM);
    float* attn_b     = alloc((size_t)E * Hh);
    float* msg_b      = alloc((size_t)E * MSG);
    float* lnbuf      = alloc((size_t)E * MSG);     // LN(msg); reused as leaky(msg@W_msg)
    float* hbuf       = alloc((size_t)E * HID_MSG); // MLP hidden; reused as GAT values

    dim3 blk(256);
    auto g2 = [](int M, int N) { return dim3((unsigned)((N + 127) / 128), (unsigned)((M + 63) / 64)); };

    // --- node projections ---------------------------------------------------
    gemm_wmma<EPI_NONE><<<g2(BN, MSG), blk, 0, stream>>>(node_repr, W_node_src, nullptr, node_msg_b, BN, MSG, DIM);
    gemm_wmma<EPI_NONE><<<g2(BN, MSG), blk, 0, stream>>>(node_repr, W_node_tgt, nullptr, node_tgt_b, BN, MSG, DIM);
    gemm_wmma<EPI_BIAS_SIGMOID><<<g2(BN, MSG), blk, 0, stream>>>(node_repr, W_gate, b_gate, gate_sig, BN, MSG, DIM);
    gemm_wmma<EPI_BIAS_SIGMOID><<<g2(BN, HD),  blk, 0, stream>>>(node_repr, W_gat_gate, b_gat_gate, gat_sig, BN, HD, DIM);

    // --- edge message -------------------------------------------------------
    gemm_wmma<EPI_NONE><<<g2(E, MSG), blk, 0, stream>>>(edge_repr, W_edge_msg, nullptr, msg_b, E, MSG, MSG);
    edge_combine_k<<<E, blk, 0, stream>>>(msg_b, node_msg_b, node_tgt_b, edge_index, mask_bw);

    // --- message MLP (LN -> GELU MLP) + mask --------------------------------
    layernorm_k<<<E, blk, 0, stream>>>(msg_b, msg_ln_g, msg_ln_b, lnbuf, MSG);
    gemm_wmma<EPI_BIAS_GELU><<<g2(E, HID_MSG), blk, 0, stream>>>(lnbuf, msg_W1, msg_b1, hbuf, E, HID_MSG, MSG);
    gemm_wmma<EPI_BIAS><<<g2(E, MSG), blk, 0, stream>>>(hbuf, msg_W2, msg_b2, msg_b, E, MSG, HID_MSG);
    mask_msg_k<<<E, blk, 0, stream>>>(msg_b, edge_mask);

    // --- global (pooled, gated) path ---------------------------------------
    pool_k<<<BN, blk, 0, stream>>>(msg_b, edge_mask, gate_sig, o_b);
    gemm_wmma<EPI_NONE><<<g2(BN, DIM), blk, 0, stream>>>(o_b, W_out, nullptr, dh1, BN, DIM, MSG);

    // --- GAT path -----------------------------------------------------------
    gemm_wmma<EPI_LEAKY><<<g2(E, MSG), blk, 0, stream>>>(msg_b, W_msg, nullptr, lnbuf, E, MSG, MSG);
    gemm_wmma<EPI_NONE><<<g2(E, Hh), blk, 0, stream>>>(lnbuf, W_attn_bias, nullptr, attn_b, E, Hh, MSG);
    softmax_k<<<(BN * Hh) / 8, blk, 0, stream>>>(attn_b, edge_mask);
    gemm_wmma<EPI_NONE><<<g2(E, HD), blk, 0, stream>>>(msg_b, W_gat_value, nullptr, hbuf, E, HD, MSG);
    gat_agg_k<<<BN, blk, 0, stream>>>(attn_b, hbuf, gat_sig, og_b);
    gemm_wmma<EPI_NONE><<<g2(BN, DIM), blk, 0, stream>>>(og_b, W_gat_out, nullptr, dh2, BN, DIM, HD);

    // --- residual + node MLP ------------------------------------------------
    int nx = BN * DIM;
    add3_k<<<(nx + 255) / 256, blk, 0, stream>>>(node_repr, dh1, dh2, x_b, nx);
    layernorm_k<<<BN, blk, 0, stream>>>(x_b, node_ln_g, node_ln_b, ln_node, DIM);
    gemm_wmma<EPI_BIAS_GELU><<<g2(BN, HID_NODE), blk, 0, stream>>>(ln_node, node_W1, node_b1, node_h, BN, HID_NODE, DIM);
    gemm_wmma<EPI_BIAS><<<g2(BN, DIM), blk, 0, stream>>>(node_h, node_W2, node_b2, mlp_out, BN, DIM, HID_NODE);
    add2_k<<<(nx + 255) / 256, blk, 0, stream>>>(x_b, mlp_out, out, nx);
}
